// TransformerEncoder_14843406975613
// MI455X (gfx1250) — compile-verified, matrix operand order checked by reading, not run on hardware
//
#include <hip/hip_runtime.h>
#include <cstdint>
#include <cstddef>

// ---------------- problem constants (from reference) ----------------
#define NLAYER 4
#define DMODEL 768
#define FFDIM  3072
#define NHEAD  12
#define SEQ    4096
#define BATCH  4
#define WIN    128
#define HDIM   64
#define TOK    (BATCH*SEQ)          // 16384 tokens
#define SPAD   (SEQ + 2*WIN)        // 4352: V columns padded by window on both sides
#define SSTR   392                  // LDS score row stride (floats): 16B-aligned, conflict-friendly

typedef __attribute__((ext_vector_type(16))) _Float16 v16h;
typedef __attribute__((ext_vector_type(8)))  _Float16 v8h;
typedef __attribute__((ext_vector_type(8)))  float    v8f;

// Load one 16x32 f16 WMMA A/B fragment. `base` points at (tile_row0, k0) of a
// row-major [rows x ld] f16 matrix. Per CDNA5 ISA 7.12.2 (wave32):
//   lane L holds row m = L&15;  element e -> k = e + (e<8?0:8) + (L<16?0:8)
// i.e. two contiguous 16-byte runs per lane at k-offsets {hi8, hi8+16}.
static __device__ __forceinline__ v16h load_frag(const _Float16* __restrict__ base,
                                                 int ld, int lane)
{
  const _Float16* p = base + (long)(lane & 15) * ld + ((lane >> 4) & 1) * 8;
  v8h lo = *(const v8h*)(p);
  v8h hi = *(const v8h*)(p + 16);
  return __builtin_shufflevector(lo, hi, 0,1,2,3,4,5,6,7,8,9,10,11,12,13,14,15);
}

static __device__ __forceinline__ v8f wmma_f32(v16h a, v16h b, v8f c)
{
  // D = A(16x32 f16) * B(32x16 f16) + C(16x16 f32)
  return __builtin_amdgcn_wmma_f32_16x16x32_f16(false, a, false, b, (short)0, c,
                                                false, false);
}

// ---------------- weight convert + transpose: W[KxN] f32 -> Wt[NxK] f16 -----
__global__ void wtrans_kernel(const float* __restrict__ W, _Float16* __restrict__ Wt,
                              int K, int N)
{
  long total = (long)K * N;
  for (long i = (long)blockIdx.x * blockDim.x + threadIdx.x; i < total;
       i += (long)gridDim.x * blockDim.x) {
    long k = i / N, n = i - k * N;
    Wt[n * (long)K + k] = (_Float16)W[i];
  }
}

// ---------------- x init: copy f32 and make f16 shadow ----------------------
__global__ void xinit_kernel(const float* __restrict__ x, float* __restrict__ xf32,
                             _Float16* __restrict__ xf16)
{
  long total = (long)TOK * DMODEL;
  for (long i = (long)blockIdx.x * blockDim.x + threadIdx.x; i < total;
       i += (long)gridDim.x * blockDim.x) {
    float v = x[i];
    xf32[i] = v;
    xf16[i] = (_Float16)v;
  }
}

// ---------------- WMMA GEMM: C = (A[MxK] * Bt[NxK]^T + bias) * scale --------
// 8 waves (2 M x 4 N), wave tile 32x64, block tile 64x256.
// out16 modes: vtmode==0 -> row-major [MxN]; vtmode==1 -> V-transposed
// layout vt[((b*NHEAD+h)*HDIM+hd)*SPAD + WIN + s]. out32: row-major f32.
__launch_bounds__(256)
__global__ void gemm_kernel(const _Float16* __restrict__ A,
                            const _Float16* __restrict__ Bt,
                            const float* __restrict__ bias,
                            _Float16* __restrict__ out16,
                            float* __restrict__ out32,
                            int M, int N, int K, float scale, int relu, int vtmode)
{
  const int lane = threadIdx.x & 31;
  const int wv   = threadIdx.x >> 5;
  const int mw   = wv & 1;       // 2 M-waves
  const int nw   = wv >> 1;      // 4 N-waves
  const long m0  = (long)blockIdx.x * 64 + mw * 32;
  const long n0  = (long)blockIdx.y * 256 + nw * 64;

  v8f acc[2][4];
#pragma unroll
  for (int i = 0; i < 2; i++)
#pragma unroll
    for (int j = 0; j < 4; j++) acc[i][j] = (v8f){0.f,0.f,0.f,0.f,0.f,0.f,0.f,0.f};

  const _Float16* Ap = A  + m0 * K;
  const _Float16* Bp = Bt + n0 * K;

  for (int kk = 0; kk < K; kk += 32) {
    v16h a0 = load_frag(Ap + kk,            K, lane);
    v16h a1 = load_frag(Ap + 16L * K + kk,  K, lane);
    v16h b0 = load_frag(Bp + kk,            K, lane);
    v16h b1 = load_frag(Bp + 16L * K + kk,  K, lane);
    v16h b2 = load_frag(Bp + 32L * K + kk,  K, lane);
    v16h b3 = load_frag(Bp + 48L * K + kk,  K, lane);
    acc[0][0] = wmma_f32(a0, b0, acc[0][0]);
    acc[0][1] = wmma_f32(a0, b1, acc[0][1]);
    acc[0][2] = wmma_f32(a0, b2, acc[0][2]);
    acc[0][3] = wmma_f32(a0, b3, acc[0][3]);
    acc[1][0] = wmma_f32(a1, b0, acc[1][0]);
    acc[1][1] = wmma_f32(a1, b1, acc[1][1]);
    acc[1][2] = wmma_f32(a1, b2, acc[1][2]);
    acc[1][3] = wmma_f32(a1, b3, acc[1][3]);
  }

  // Epilogue. C/D layout: lane L col n = L&15; VGPR r row m = r + 8*(L>=16).
  const int hi = (lane >> 4) & 1;
  const int nl = lane & 15;
#pragma unroll
  for (int i = 0; i < 2; i++) {
#pragma unroll
    for (int j = 0; j < 4; j++) {
      long n = n0 + j * 16 + nl;
      float bv = bias ? bias[n] : 0.f;
#pragma unroll
      for (int r = 0; r < 8; r++) {
        long m = m0 + i * 16 + r + hi * 8;
        float v = (acc[i][j][r] + bv) * scale;
        if (relu) v = fmaxf(v, 0.f);
        if (out32) out32[m * N + n] = v;
        if (out16) {
          if (vtmode) {
            long b = m >> 12, s = m & (SEQ - 1);     // SEQ = 4096
            long h = n >> 6,  hd = n & (HDIM - 1);
            out16[((b * NHEAD + h) * HDIM + hd) * SPAD + WIN + s] = (_Float16)v;
          } else {
            out16[m * N + n] = (_Float16)v;
          }
        }
      }
    }
  }
}

// ---------------- sliding-window attention --------------------------------
// One block per (b, h, 64-query slab). 4 waves; wave owns 16 queries.
// Phase 1: S = Q*K^T (masked) -> LDS f32 [64 x SSTR]
// Phase 2: per-row softmax -> P f16 in LDS [64 x SSTR]
// Phase 3: O = P*V via WMMA from zero-padded transposed V; f32 out.
__launch_bounds__(128)
__global__ void attn_kernel(const _Float16* __restrict__ q16,
                            const _Float16* __restrict__ k16,   // padded: +WIN rows both ends
                            const _Float16* __restrict__ vt,    // [B][H][HDIM][SPAD]
                            float* __restrict__ outf)           // [TOK x DMODEL]
{
  extern __shared__ float smem[];
  float*    Sm = smem;                               // 64 x SSTR f32
  _Float16* Pm = (_Float16*)(smem + 64 * SSTR);      // 64 x SSTR f16

  const int lane = threadIdx.x & 31;
  const int wv   = threadIdx.x >> 5;
  const int hi   = (lane >> 4) & 1;
  const int nl   = lane & 15;
  const int qb   = blockIdx.x * 64;
  const int h    = blockIdx.y;
  const int b    = blockIdx.z;
  const int c    = qb >> 7;            // window chunk (width WIN=128)
  const int kb   = c * WIN - WIN;      // global key pos of window element kj=0

  // --- Phase 1: scores ---
  const _Float16* Aq = q16 + ((long)(b * SEQ + qb + wv * 16)) * DMODEL + h * HDIM;
  v16h a_lo = load_frag(Aq,      DMODEL, lane);
  v16h a_hi = load_frag(Aq + 32, DMODEL, lane);
  const _Float16* Kb = k16 + ((long)(b * SEQ + kb) + WIN) * DMODEL + h * HDIM;

  for (int nt = 0; nt < 24; nt++) {                  // 24 key tiles of 16 -> 384 keys
    v8f s = (v8f){0.f,0.f,0.f,0.f,0.f,0.f,0.f,0.f};
    v16h kb0 = load_frag(Kb + (long)nt * 16 * DMODEL,      DMODEL, lane);
    v16h kb1 = load_frag(Kb + (long)nt * 16 * DMODEL + 32, DMODEL, lane);
    s = wmma_f32(a_lo, kb0, s);
    s = wmma_f32(a_hi, kb1, s);
    int kj   = nt * 16 + nl;
    int kpos = kb + kj;
#pragma unroll
    for (int r = 0; r < 8; r++) {
      int ml  = wv * 16 + r + hi * 8;                // local query row 0..63
      int qi  = (qb + ml) & (WIN - 1);               // index within window chunk
      int rel = kj - qi - WIN;
      bool ok = (rel >= -WIN) && (rel <= WIN) && (kpos >= 0) && (kpos < SEQ);
      Sm[ml * SSTR + kj] = ok ? s[r] : -1e30f;
    }
  }
  __syncthreads();

  // --- Phase 2: softmax (one thread per query row) ---
  if (threadIdx.x < 64) {
    float* Sr = Sm + threadIdx.x * SSTR;
    _Float16* Pr = Pm + threadIdx.x * SSTR;
    float mx = -1e30f;
    for (int j = 0; j < 384; j++) mx = fmaxf(mx, Sr[j]);
    float sum = 0.f;
    for (int j = 0; j < 384; j++) {
      float e = __expf(Sr[j] - mx);
      sum += e;
      Pr[j] = (_Float16)e;
    }
    float inv = 1.f / sum;
    for (int j = 0; j < 384; j++) Pr[j] = (_Float16)((float)Pr[j] * inv);
  }
  __syncthreads();

  // --- Phase 3: O = P * V ---
  v8f o[4];
#pragma unroll
  for (int j = 0; j < 4; j++) o[j] = (v8f){0.f,0.f,0.f,0.f,0.f,0.f,0.f,0.f};

  const _Float16* Vb = vt + ((long)(b * NHEAD + h) * HDIM) * SPAD + (WIN + kb);
  for (int kc = 0; kc < 12; kc++) {                  // 12 chunks of 32 keys
    v16h ap = load_frag(Pm + (long)wv * 16 * SSTR + kc * 32, SSTR, lane);
#pragma unroll
    for (int j = 0; j < 4; j++) {
      v16h bv = load_frag(Vb + (long)(j * 16) * SPAD + kc * 32, SPAD, lane);
      o[j] = wmma_f32(ap, bv, o[j]);
    }
  }
#pragma unroll
  for (int j = 0; j < 4; j++) {
    long n = h * HDIM + j * 16 + nl;
#pragma unroll
    for (int r = 0; r < 8; r++) {
      long m = (long)b * SEQ + qb + wv * 16 + r + hi * 8;
      outf[m * DMODEL + n] = o[j][r];
    }
  }
}

// ---------------- fused residual-add + LayerNorm --------------------------
__launch_bounds__(256)
__global__ void add_ln_kernel(const float* __restrict__ xin,
                              const float* __restrict__ dres,
                              const float* __restrict__ g,
                              const float* __restrict__ be,
                              float* __restrict__ out32,
                              _Float16* __restrict__ out16)
{
  __shared__ float red[256];
  const long row = blockIdx.x;
  const int  t   = threadIdx.x;
  const float* xr = xin  + row * DMODEL;
  const float* dr = dres + row * DMODEL;

  float v[3];
#pragma unroll
  for (int i = 0; i < 3; i++) v[i] = xr[t + i * 256] + dr[t + i * 256];

  red[t] = v[0] + v[1] + v[2];
  __syncthreads();
  for (int off = 128; off > 0; off >>= 1) {
    if (t < off) red[t] += red[t + off];
    __syncthreads();
  }
  float mean = red[0] * (1.f / DMODEL);
  __syncthreads();

  float d0 = v[0] - mean, d1 = v[1] - mean, d2 = v[2] - mean;
  red[t] = d0 * d0 + d1 * d1 + d2 * d2;
  __syncthreads();
  for (int off = 128; off > 0; off >>= 1) {
    if (t < off) red[t] += red[t + off];
    __syncthreads();
  }
  float rs = rsqrtf(red[0] * (1.f / DMODEL) + 1e-5f);

#pragma unroll
  for (int i = 0; i < 3; i++) {
    int col = t + i * 256;
    float y = (v[i] - mean) * rs * g[col] + be[col];
    out32[row * DMODEL + col] = y;
    out16[row * DMODEL + col] = (_Float16)y;
  }
}

// ---------------- host side ------------------------------------------------
extern "C" void kernel_launch(void* const* d_in, const int* in_sizes, int n_in,
                              void* d_out, int out_size, void* d_ws, size_t ws_size,
                              hipStream_t stream)
{
  const float* x   = (const float*)d_in[0];
  const float* Wq  = (const float*)d_in[1];
  const float* bq  = (const float*)d_in[2];
  const float* Wk  = (const float*)d_in[3];
  const float* bk  = (const float*)d_in[4];
  const float* Wv  = (const float*)d_in[5];
  const float* bv  = (const float*)d_in[6];
  const float* W1  = (const float*)d_in[7];
  const float* b1  = (const float*)d_in[8];
  const float* W2  = (const float*)d_in[9];
  const float* b2  = (const float*)d_in[10];
  const float* g1  = (const float*)d_in[11];
  const float* be1 = (const float*)d_in[12];
  const float* g2  = (const float*)d_in[13];
  const float* be2 = (const float*)d_in[14];

  char* ws = (char*)d_ws;
  size_t off = 0;
  auto alloc = [&](size_t bytes) -> char* {
    char* p = ws + off;
    off += (bytes + 255) & ~(size_t)255;
    return p;
  };

  const size_t szAttnW = (size_t)NLAYER * DMODEL * DMODEL * sizeof(_Float16);
  const size_t szFfnW  = (size_t)NLAYER * DMODEL * FFDIM * sizeof(_Float16);
  const size_t szK16   = (size_t)(TOK + 2 * WIN) * DMODEL * sizeof(_Float16);
  const size_t szVt    = (size_t)BATCH * NHEAD * HDIM * SPAD * sizeof(_Float16);

  _Float16* wqt  = (_Float16*)alloc(szAttnW);
  _Float16* wkt  = (_Float16*)alloc(szAttnW);
  _Float16* wvt  = (_Float16*)alloc(szAttnW);
  _Float16* w1t  = (_Float16*)alloc(szFfnW);
  _Float16* w2t  = (_Float16*)alloc(szFfnW);
  float*    xf32 = (float*)   alloc((size_t)TOK * DMODEL * sizeof(float));
  _Float16* xf16 = (_Float16*)alloc((size_t)TOK * DMODEL * sizeof(_Float16));
  _Float16* q16  = (_Float16*)alloc((size_t)TOK * DMODEL * sizeof(_Float16));
  _Float16* k16  = (_Float16*)alloc(szK16);
  _Float16* vt   = (_Float16*)alloc(szVt);
  _Float16* h16  = (_Float16*)alloc((size_t)TOK * FFDIM * sizeof(_Float16));
  float*    resb = (float*)   alloc((size_t)TOK * DMODEL * sizeof(float));
  (void)off; (void)ws_size; // ~356 MB total workspace used

  // Zero the padded K/V buffers once per launch (pads are never written and
  // must be finite/zero so masked window positions contribute nothing).
  hipMemsetAsync(k16, 0, szK16, stream);
  hipMemsetAsync(vt,  0, szVt,  stream);

  // Convert+transpose all weights to f16 (so every WMMA operand is loaded
  // with two contiguous 16B reads per lane).
  for (int l = 0; l < NLAYER; l++) {
    size_t dd = (size_t)l * DMODEL * DMODEL;
    size_t df = (size_t)l * DMODEL * FFDIM;
    wtrans_kernel<<<1024, 256, 0, stream>>>(Wq + dd, wqt + dd, DMODEL, DMODEL);
    wtrans_kernel<<<1024, 256, 0, stream>>>(Wk + dd, wkt + dd, DMODEL, DMODEL);
    wtrans_kernel<<<1024, 256, 0, stream>>>(Wv + dd, wvt + dd, DMODEL, DMODEL);
    wtrans_kernel<<<2048, 256, 0, stream>>>(W1 + df, w1t + df, DMODEL, FFDIM);
    wtrans_kernel<<<2048, 256, 0, stream>>>(W2 + df, w2t + df, FFDIM, DMODEL);
  }

  xinit_kernel<<<2048, 256, 0, stream>>>(x, xf32, xf16);

  const dim3 gD (TOK / 64, DMODEL / 256);  // GEMM with N = 768
  const dim3 gFF(TOK / 64, FFDIM / 256);   // GEMM with N = 3072
  const dim3 gAt(SEQ / 64, NHEAD, BATCH);
  const size_t attn_lds = (size_t)64 * SSTR * sizeof(float)
                        + (size_t)64 * SSTR * sizeof(_Float16); // 150528 B
  const float qscale = 0.125f;             // 1/sqrt(HDIM)

  for (int l = 0; l < NLAYER; l++) {
    size_t dd = (size_t)l * DMODEL * DMODEL;
    size_t df = (size_t)l * DMODEL * FFDIM;

    // Q/K/V projections
    gemm_kernel<<<gD, 256, 0, stream>>>(xf16, wqt + dd, bq + (size_t)l * DMODEL,
                                        q16, nullptr, TOK, DMODEL, DMODEL,
                                        qscale, 0, 0);
    gemm_kernel<<<gD, 256, 0, stream>>>(xf16, wkt + dd, bk + (size_t)l * DMODEL,
                                        k16 + (size_t)WIN * DMODEL, nullptr,
                                        TOK, DMODEL, DMODEL, 1.f, 0, 0);
    gemm_kernel<<<gD, 256, 0, stream>>>(xf16, wvt + dd, bv + (size_t)l * DMODEL,
                                        vt, nullptr, TOK, DMODEL, DMODEL,
                                        1.f, 0, 1 /*V-transposed*/);

    attn_kernel<<<gAt, 128, attn_lds, stream>>>(q16, k16, vt, resb);

    add_ln_kernel<<<TOK, 256, 0, stream>>>(xf32, resb,
                                           g1 + (size_t)l * DMODEL,
                                           be1 + (size_t)l * DMODEL, xf32, xf16);

    // FFN
    gemm_kernel<<<gFF, 256, 0, stream>>>(xf16, w1t + df, b1 + (size_t)l * FFDIM,
                                         h16, nullptr, TOK, FFDIM, DMODEL,
                                         1.f, 1 /*relu*/, 0);
    gemm_kernel<<<gD, 256, 0, stream>>>(h16, w2t + df, b2 + (size_t)l * DMODEL,
                                        nullptr, resb, TOK, DMODEL, FFDIM,
                                        1.f, 0, 0);

    add_ln_kernel<<<TOK, 256, 0, stream>>>(xf32, resb,
                                           g2 + (size_t)l * DMODEL,
                                           be2 + (size_t)l * DMODEL, xf32, xf16);
  }

  hipMemcpyAsync(d_out, xf32, (size_t)TOK * DMODEL * sizeof(float),
                 hipMemcpyDeviceToDevice, stream);
}
